// RPNmodel_987842478834
// MI455X (gfx1250) — matrix operand order repaired and checked by reading, not compiled, not run on hardware
//
#include <hip/hip_runtime.h>
#include <hip/hip_bf16.h>
#include <math.h>

typedef __attribute__((ext_vector_type(16))) _Float16 v16h;
typedef __attribute__((ext_vector_type(8)))  _Float16 v8h;
typedef __attribute__((ext_vector_type(8)))  float    v8f;
typedef __attribute__((ext_vector_type(4)))  unsigned int u32x4;
typedef __attribute__((ext_vector_type(8)))  unsigned int u32x8;

#define FEAT     52
#define M_SPAT   2704
#define M_PAD    2720
#define MT_N     170        // M tiles of 16
#define K3X3     2304       // 256*9
#define KK3      72         // K steps of 32 for 3x3 conv
#define KK3H     36         // half of the K steps (per TDM chunk)
#define NT3      16         // 256/16 output-channel tiles
#define KK1      8          // 256/32 K steps for 1x1 convs
#define NT1      4          // 64/16 output-channel tiles (54 padded to 64)
#define NANCH    24336
#define NSORT    32768
#define PRE_NMS_N 12000
#define POST_NMS_N 2000
#define NWORDS   375        // 12000/32

// float offsets into d_out (rois, locs, cls, obj, cls)
#define LOCS_OFF  8000
#define CLS1_OFF  105344
#define OBJ_OFF   154016
#define CLS2_OFF  178352

// ---------------------------------------------------------------------------
// Pack 3x3 conv weights (OIHW, fp32) into B-fragment order, f16.
// B fragment (16x16x32 f16): lane&15 = N; K = e + (e>=8?8:0) + (lane>=16?8:0)
// ---------------------------------------------------------------------------
__global__ void pack_w3(const float* __restrict__ w, _Float16* __restrict__ Bp) {
    int id = blockIdx.x * blockDim.x + threadIdx.x;          // < 16*72*32*16
    if (id >= NT3 * KK3 * 32 * 16) return;
    int e    = id & 15;
    int lane = (id >> 4) & 31;
    int kk   = (id >> 9) % KK3;
    int nt   = id / (16 * 32 * KK3);
    int n = nt * 16 + (lane & 15);
    int k = kk * 32 + e + ((e >= 8) ? 8 : 0) + ((lane >= 16) ? 8 : 0);
    Bp[id] = (_Float16)w[(size_t)n * K3X3 + k];
}

// ---------------------------------------------------------------------------
// Pack combined 1x1 weights (reg 36ch + cls 18ch -> 54, pad to 64), f16.
// ---------------------------------------------------------------------------
__global__ void pack_w1(const float* __restrict__ wr, const float* __restrict__ wc,
                        _Float16* __restrict__ Bp) {
    int id = blockIdx.x * blockDim.x + threadIdx.x;          // < 4*8*32*16
    if (id >= NT1 * KK1 * 32 * 16) return;
    int e    = id & 15;
    int lane = (id >> 4) & 31;
    int kk   = (id >> 9) & 7;
    int nt   = id >> 12;
    int n = nt * 16 + (lane & 15);
    int k = kk * 32 + e + ((e >= 8) ? 8 : 0) + ((lane >= 16) ? 8 : 0);
    float v = 0.f;
    if (n < 36)      v = wr[(size_t)n * 256 + k];
    else if (n < 54) v = wc[(size_t)(n - 36) * 256 + k];
    Bp[id] = (_Float16)v;
}

// ---------------------------------------------------------------------------
// im2col of the padded 3x3 window, written directly in A-fragment order, f16.
// A fragment (16x32 f16): lane&15 = M row; K = e + (e>=8?8:0) + (lane>=16?8:0)
// K index = c*9 + (ky*3+kx)
// ---------------------------------------------------------------------------
__global__ void pack_im2col(const float* __restrict__ x, _Float16* __restrict__ Ap) {
    int id = blockIdx.x * blockDim.x + threadIdx.x;          // < 170*72*32*16
    if (id >= MT_N * KK3 * 32 * 16) return;
    int e    = id & 15;
    int lane = (id >> 4) & 31;
    int kk   = (id >> 9) % KK3;
    int mt   = id / (16 * 32 * KK3);
    int k = kk * 32 + e + ((e >= 8) ? 8 : 0) + ((lane >= 16) ? 8 : 0);
    int c = k / 9, r = k % 9;
    int m = mt * 16 + (lane & 15);
    float v = 0.f;
    if (m < M_SPAT) {
        int y = m / FEAT, xx = m % FEAT;
        int iy = y + r / 3 - 1, ix = xx + r % 3 - 1;
        if (iy >= 0 && iy < FEAT && ix >= 0 && ix < FEAT)
            v = x[(size_t)c * (FEAT * FEAT) + iy * FEAT + ix];
    }
    Ap[id] = (_Float16)v;
}

// ---------------------------------------------------------------------------
// 3x3 conv as implicit GEMM. 8 waves/block share one M-tile: the A slab is
// DMA'd into LDS once per block via the Tensor Data Mover (two 36 KB chunks),
// then all waves read A fragments from LDS (ds_load) while streaming their
// private B fragments from L2. D# built in SGPRs per CDNA5 ISA.
// ---------------------------------------------------------------------------
__global__ void conv3_wmma(const _Float16* __restrict__ Apack,
                           const _Float16* __restrict__ Bpack,
                           const float* __restrict__ bias,
                           _Float16* __restrict__ hbuf) {
    __shared__ alignas(16) _Float16 shA[KK3H * 32 * 16];     // 36 KB chunk
    int lane = threadIdx.x & 31;
    int wv   = threadIdx.x >> 5;
    int mt = blockIdx.x >> 1;                 // 0..169 (shared by whole block)
    int nt = ((blockIdx.x & 1) << 3) + wv;    // 0..15
    const char* Aslab = (const char*)(Apack + (size_t)mt * KK3 * 32 * 16);
    unsigned int ldsoff = (unsigned int)(unsigned long long)&shA[0];
    const v16h* Bp = (const v16h*)Bpack + (size_t)nt * KK3 * 32;
    v8f acc = {};
    const unsigned int CHUNK_B  = KK3H * 32 * 16 * 2;        // 36864 bytes
    const unsigned int CHUNK_E8 = CHUNK_B / 8;               // 4608 8-byte elems
    for (int half = 0; half < 2; ++half) {
        if (wv == 0) {
            unsigned long long ga =
                (unsigned long long)Aslab + (unsigned long long)half * CHUNK_B;
            u32x4 g0;                         // D# group 0 (count/lds/global)
            g0[0] = 1u;                                       // count=1
            g0[1] = ldsoff;                                   // lds_addr
            g0[2] = (unsigned int)ga;                         // global_addr lo
            g0[3] = ((unsigned int)(ga >> 32) & 0x01FFFFFFu)  // global_addr hi
                    | 0x80000000u;                            // type=2 (image)
            u32x8 g1;                         // D# group 1 (dims/tile/stride)
            g1[0] = 3u << 16;                                 // data_size=8B
            g1[1] = (CHUNK_E8 & 0xFFFFu) << 16;               // tensor_dim0 lo16
            g1[2] = (CHUNK_E8 >> 16) | (1u << 16);            // dim0 hi | dim1=1
            g1[3] = (CHUNK_E8 & 0xFFFFu) << 16;               // tile_dim0
            g1[4] = 0u;                                       // tile_dim1/2 = 0
            g1[5] = CHUNK_E8;                                 // dim0_stride lo
            g1[6] = 0u;
            g1[7] = 0u;
            asm volatile("tensor_load_to_lds %0, %1"
                         :: "s"(g0), "s"(g1) : "memory");
            __builtin_amdgcn_s_wait_tensorcnt(0);
        }
        __syncthreads();                       // LDS chunk visible to all waves
        const v16h* As = (const v16h*)shA;
        for (int kk2 = 0; kk2 < KK3H; ++kk2) {
            int kk = half * KK3H + kk2;
            v16h a = As[kk2 * 32 + lane];                     // ds_load_b128 x2
            v16h b = Bp[kk * 32 + lane];                      // global b128 x2
            __builtin_prefetch(Bp + (kk + 1) * 32 + lane, 0, 3);
            acc = __builtin_amdgcn_wmma_f32_16x16x32_f16(
                false, a, false, b, (short)0, acc, false, false);
        }
        __syncthreads();                       // done reading before next DMA
    }
    // C layout: lane<16 -> N=lane, M=r ; lane>=16 -> N=lane-16, M=8+r
    int n = nt * 16 + (lane & 15);
    float bv = bias[n];
    int mbase = mt * 16 + ((lane >> 4) << 3);
#pragma unroll
    for (int r = 0; r < 8; ++r)
        hbuf[(size_t)(mbase + r) * 256 + n] = (_Float16)(acc[r] + bv);
}

// ---------------------------------------------------------------------------
// Fused 1x1 convs (reg+cls) as GEMM: M=2720, N=64, K=256. Writes locs /
// cls_scores (twice) / obj straight into d_out at their flat offsets.
// ---------------------------------------------------------------------------
__global__ void conv1x1_wmma(const _Float16* __restrict__ hbuf,
                             const _Float16* __restrict__ Bpack1,
                             const float* __restrict__ reg_b,
                             const float* __restrict__ cls_b,
                             float* __restrict__ outf) {
    int wid  = blockIdx.x * (blockDim.x >> 5) + (threadIdx.x >> 5);
    int lane = threadIdx.x & 31;
    int mt = wid >> 2;       // 0..169
    int nt = wid & 3;        // 0..3
    int mrow = mt * 16 + (lane & 15);
    const _Float16* hrow = hbuf + (size_t)mrow * 256;
    int kshift = (lane >= 16) ? 8 : 0;
    const v16h* Bp = (const v16h*)Bpack1 + (size_t)nt * KK1 * 32;
    v8f acc = {};
    for (int kk = 0; kk < KK1; ++kk) {
        const _Float16* p = hrow + kk * 32 + kshift;
        v8h lo = *(const v8h*)p;          // K = base .. base+7
        v8h hi = *(const v8h*)(p + 16);   // K = base+16 .. base+23
        v16h a;
#pragma unroll
        for (int q = 0; q < 8; ++q) { a[q] = lo[q]; a[q + 8] = hi[q]; }
        v16h b = Bp[kk * 32 + lane];
        acc = __builtin_amdgcn_wmma_f32_16x16x32_f16(
            false, a, false, b, (short)0, acc, false, false);
    }
    int n = nt * 16 + (lane & 15);
    float bv = (n < 36) ? reg_b[n] : ((n < 54) ? cls_b[n - 36] : 0.f);
    int mbase = mt * 16 + ((lane >> 4) << 3);
#pragma unroll
    for (int r = 0; r < 8; ++r) {
        int m = mbase + r;
        if (m >= M_SPAT) continue;
        float v = acc[r] + bv;
        if (n < 36) {
            outf[LOCS_OFF + (size_t)m * 36 + n] = v;
        } else if (n < 54) {
            int c = n - 36;
            outf[CLS1_OFF + (size_t)m * 18 + c] = v;
            outf[CLS2_OFF + (size_t)m * 18 + c] = v;
            if (c & 1) outf[OBJ_OFF + (size_t)m * 9 + (c >> 1)] = v;
        }
    }
}

// ---------------------------------------------------------------------------
// Box decode + clip + min-size filter + sortable 64-bit key build.
// key = (~monotone(score) << 32) | index  -> ascending sort == score desc,
// index asc (matches stable argsort(-score)). Pads sort last.
// ---------------------------------------------------------------------------
__global__ void decode_boxes(const float* __restrict__ outf,
                             float* __restrict__ boxes,
                             int* __restrict__ valid,
                             unsigned long long* __restrict__ key) {
    int i = blockIdx.x * blockDim.x + threadIdx.x;
    if (i >= NSORT) return;
    if (i >= NANCH) { key[i] = ~0ull; return; }
    int m = i / 9, a = i % 9;
    int ri = a / 3, si = a % 3;
    float ratio = (ri == 0) ? 0.5f : ((ri == 1) ? 1.0f : 2.0f);
    float scale = (si == 0) ? 4.0f : ((si == 1) ? 8.0f : 16.0f);
    float ah = 4.f * scale * sqrtf(ratio);
    float aw = 4.f * scale * sqrtf(1.f / ratio);
    float acy = (float)((m / FEAT) * 4 + 2);
    float acx = (float)((m % FEAT) * 4 + 2);
    const float* l = outf + LOCS_OFF + (size_t)i * 4;
    float cy = l[0] * ah + acy;
    float cx = l[1] * aw + acx;
    float hh = expf(l[2]) * ah;
    float ww = expf(l[3]) * aw;
    float y1 = fminf(fmaxf(cy - 0.5f * hh, 0.f), 210.f);
    float x1 = fminf(fmaxf(cx - 0.5f * ww, 0.f), 210.f);
    float y2 = fminf(fmaxf(cy + 0.5f * hh, 0.f), 210.f);
    float x2 = fminf(fmaxf(cx + 0.5f * ww, 0.f), 210.f);
    int ok = ((y2 - y1) >= 16.f) && ((x2 - x1) >= 16.f);
    float s = ok ? outf[OBJ_OFF + i] : -INFINITY;
    boxes[(size_t)i * 4 + 0] = y1;
    boxes[(size_t)i * 4 + 1] = x1;
    boxes[(size_t)i * 4 + 2] = y2;
    boxes[(size_t)i * 4 + 3] = x2;
    valid[i] = ok;
    unsigned int u  = __float_as_uint(s);
    unsigned int mo = (u & 0x80000000u) ? ~u : (u | 0x80000000u); // ascending map
    unsigned int ks = ~mo;                                        // descending
    key[i] = ((unsigned long long)ks << 32) | (unsigned int)i;
}

// one bitonic compare-exchange pass over global keys
__global__ void bitonic_step(unsigned long long* __restrict__ key, int j, int k) {
    int i = blockIdx.x * blockDim.x + threadIdx.x;
    int ixj = i ^ j;
    if (ixj > i) {
        unsigned long long a = key[i], b = key[ixj];
        bool up = ((i & k) == 0);
        if (up ? (a > b) : (a < b)) { key[i] = b; key[ixj] = a; }
    }
}

// gather top-12000 boxes/validity in sorted order
__global__ void gather_sorted(const unsigned long long* __restrict__ key,
                              const float* __restrict__ boxes,
                              const int* __restrict__ valid,
                              float* __restrict__ sboxes,
                              int* __restrict__ svalid) {
    int t = blockIdx.x * blockDim.x + threadIdx.x;
    if (t >= PRE_NMS_N) return;
    unsigned int idx = (unsigned int)(key[t] & 0xffffffffu);
    if (idx < NANCH) {
        svalid[t] = valid[idx];
#pragma unroll
        for (int q = 0; q < 4; ++q)
            sboxes[(size_t)t * 4 + q] = boxes[(size_t)idx * 4 + q];
    } else {
        svalid[t] = 0;
#pragma unroll
        for (int q = 0; q < 4; ++q) sboxes[(size_t)t * 4 + q] = 0.f;
    }
}

// ---------------------------------------------------------------------------
// Tiled IoU bitmask: one wave per 32x32 (i-chunk, j-chunk) tile, j-boxes
// staged in LDS. mask[i][cj] bit b set iff IoU(i, cj*32+b) > 0.7 and j > i.
// ---------------------------------------------------------------------------
__global__ void iou_mask(const float* __restrict__ sb, unsigned int* __restrict__ mask) {
    int cj = blockIdx.x, ci = blockIdx.y;
    int t = threadIdx.x;                       // 0..31, one wave
    __shared__ float jy1[32], jx1[32], jy2[32], jx2[32], jar[32];
    int i = ci * 32 + t;
    if (cj >= ci) {
        int j = cj * 32 + t;
        float b0 = sb[(size_t)j * 4 + 0], b1 = sb[(size_t)j * 4 + 1];
        float b2 = sb[(size_t)j * 4 + 2], b3 = sb[(size_t)j * 4 + 3];
        jy1[t] = b0; jx1[t] = b1; jy2[t] = b2; jx2[t] = b3;
        jar[t] = (b3 - b1 + 1.f) * (b2 - b0 + 1.f);
    }
    __syncthreads();
    unsigned int word = 0;
    if (cj >= ci) {
        float y1i = sb[(size_t)i * 4 + 0], x1i = sb[(size_t)i * 4 + 1];
        float y2i = sb[(size_t)i * 4 + 2], x2i = sb[(size_t)i * 4 + 3];
        float ai = (x2i - x1i + 1.f) * (y2i - y1i + 1.f);
        for (int b = 0; b < 32; ++b) {
            int j = cj * 32 + b;
            if (j <= i) continue;
            float yy1 = fmaxf(y1i, jy1[b]);
            float xx1 = fmaxf(x1i, jx1[b]);
            float yy2 = fminf(y2i, jy2[b]);
            float xx2 = fminf(x2i, jx2[b]);
            float inter = fmaxf(0.f, xx2 - xx1 + 1.f) * fmaxf(0.f, yy2 - yy1 + 1.f);
            float iou = inter / (ai + jar[b] - inter);
            if (iou > 0.7f) word |= (1u << b);
        }
    }
    mask[(size_t)i * NWORDS + cj] = word;
}

// single-workgroup sequential reduction of the bitmask (the only serial part)
__global__ void nms_seq(const int* __restrict__ svalid,
                        const unsigned int* __restrict__ mask,
                        unsigned int* __restrict__ keep) {
    __shared__ unsigned int supp[NWORDS];
    int t = threadIdx.x;
    if (t < NWORDS) {
        unsigned int w0 = 0;
        for (int b = 0; b < 32; ++b)
            if (!svalid[t * 32 + b]) w0 |= (1u << b);   // invalid start suppressed
        supp[t] = w0;
    }
    __syncthreads();
    for (int c = 0; c < NWORDS; ++c) {
        __syncthreads();
        unsigned int w = supp[c];                       // uniform across threads
        for (int b = 0; b < 32; ++b) {
            if (((w >> b) & 1u) == 0u) {                // box i kept -> suppress
                int i = c * 32 + b;
                const unsigned int* row = mask + (size_t)i * NWORDS;
                w |= row[c];                            // same-chunk updates
                if (t < NWORDS && t > c) supp[t] |= row[t];
            }
        }
        if (t == 0) supp[c] = w;
        __syncthreads();
    }
    __syncthreads();
    if (t < NWORDS) keep[t] = ~supp[t];
}

// prefix-sum ranks + write rois (zero-padded to 2000)
__global__ void compact_rois(const unsigned int* __restrict__ keep,
                             const float* __restrict__ sboxes,
                             float* __restrict__ outf) {
    __shared__ unsigned int cnt[NWORDS + 1];
    int t = threadIdx.x;
    if (t < NWORDS) cnt[t] = __popc(keep[t]);
    __syncthreads();
    if (t == 0) {
        unsigned int run = 0;
        for (int i = 0; i < NWORDS; ++i) { unsigned int v = cnt[i]; cnt[i] = run; run += v; }
        cnt[NWORDS] = run;
    }
    __syncthreads();
    unsigned int total = cnt[NWORDS];
    if (total > POST_NMS_N) total = POST_NMS_N;
    for (int idx = (int)total * 4 + t; idx < POST_NMS_N * 4; idx += blockDim.x)
        outf[idx] = 0.f;                                // zero-pad tail, no overlap
    if (t < NWORDS) {
        unsigned int word = keep[t];
        unsigned int base = cnt[t];
        for (int b = 0; b < 32; ++b) {
            if ((word >> b) & 1u) {
                if (base < POST_NMS_N) {
                    int i = t * 32 + b;
#pragma unroll
                    for (int q = 0; q < 4; ++q)
                        outf[(size_t)base * 4 + q] = sboxes[(size_t)i * 4 + q];
                }
                ++base;
            }
        }
    }
}

extern "C" void kernel_launch(void* const* d_in, const int* in_sizes, int n_in,
                              void* d_out, int out_size, void* d_ws, size_t ws_size,
                              hipStream_t stream) {
    (void)in_sizes; (void)n_in; (void)out_size; (void)ws_size;
    const float* xin = (const float*)d_in[0];
    const float* w3  = (const float*)d_in[1];
    const float* b3  = (const float*)d_in[2];
    const float* wr  = (const float*)d_in[3];
    const float* br  = (const float*)d_in[4];
    const float* wc  = (const float*)d_in[5];
    const float* bc  = (const float*)d_in[6];
    float* outf = (float*)d_out;

    char* ws = (char*)d_ws;
    size_t off = 0;
    auto carve = [&](size_t bytes) -> char* {
        char* p = ws + off;
        off = (off + bytes + 255) & ~(size_t)255;
        return p;
    };
    _Float16* Apack  = (_Float16*)carve((size_t)MT_N * KK3 * 32 * 16 * 2);   // 12.5 MB
    _Float16* Bpack3 = (_Float16*)carve((size_t)NT3 * KK3 * 32 * 16 * 2);    // 1.13 MB
    _Float16* hbuf   = (_Float16*)carve((size_t)M_PAD * 256 * 2);            // 1.4 MB
    _Float16* Bpack1 = (_Float16*)carve((size_t)NT1 * KK1 * 32 * 16 * 2);    // 32 KB
    float*    boxesw = (float*)carve((size_t)NANCH * 4 * 4);
    int*      validw = (int*)carve((size_t)NANCH * 4);
    unsigned long long* keyw = (unsigned long long*)carve((size_t)NSORT * 8);
    float*    sboxes = (float*)carve((size_t)PRE_NMS_N * 4 * 4);
    int*      svalid = (int*)carve((size_t)PRE_NMS_N * 4);
    unsigned int* maskw = (unsigned int*)carve((size_t)PRE_NMS_N * NWORDS * 4); // 18 MB
    unsigned int* keepw = (unsigned int*)carve((size_t)NWORDS * 4);

    // weight / im2col packing (fp32 -> f16 in WMMA fragment order)
    pack_w3<<<(NT3 * KK3 * 32 * 16) / 256, 256, 0, stream>>>(w3, Bpack3);
    pack_w1<<<(NT1 * KK1 * 32 * 16) / 256, 256, 0, stream>>>(wr, wc, Bpack1);
    pack_im2col<<<(MT_N * KK3 * 32 * 16) / 256, 256, 0, stream>>>(xin, Apack);

    // GEMMs on WMMA: 3x3 conv (TDM->LDS A staging, 2720 waves), fused 1x1 convs
    conv3_wmma<<<MT_N * 2, 256, 0, stream>>>(Apack, Bpack3, b3, hbuf);
    conv1x1_wmma<<<(MT_N * NT1) / 8, 256, 0, stream>>>(hbuf, Bpack1, br, bc, outf);

    // decode + key build, then global bitonic sort of 32768 64-bit keys
    decode_boxes<<<NSORT / 256, 256, 0, stream>>>(outf, boxesw, validw, keyw);
    for (int k = 2; k <= NSORT; k <<= 1)
        for (int j = k >> 1; j > 0; j >>= 1)
            bitonic_step<<<NSORT / 256, 256, 0, stream>>>(keyw, j, k);

    gather_sorted<<<(PRE_NMS_N + 255) / 256, 256, 0, stream>>>(keyw, boxesw, validw,
                                                               sboxes, svalid);
    // tiled parallel IoU bitmask, one-block sequential reduction, compact
    iou_mask<<<dim3(NWORDS, NWORDS), 32, 0, stream>>>(sboxes, maskw);
    nms_seq<<<1, 384, 0, stream>>>(svalid, maskw, keepw);
    compact_rois<<<1, 384, 0, stream>>>(keepw, sboxes, outf);
}
